// Transducer_2757369004491
// MI455X (gfx1250) — matrix operand order repaired and checked by reading, not compiled, not run on hardware
//
#include <hip/hip_runtime.h>
#include <hip/hip_bf16.h>
#include <math.h>

typedef __attribute__((ext_vector_type(16))) __bf16 v16bf;
typedef __attribute__((ext_vector_type(8)))  float  v8f;

#define DE 640
#define HID 640
#define VOC 1024
#define BATCH 4
#define TT 160
#define UU 80
#define MBATCH (TT*UU)   // 12800 rows per batch

// Native CDNA5 V_TANH_F32 (single trans op) when available; OCML fallback.
#if __has_builtin(__builtin_amdgcn_tanhf)
  #define FAST_TANH(x) __builtin_amdgcn_tanhf(x)
#else
  #define FAST_TANH(x) tanhf(x)
#endif

// POD 16-byte chunk so it can live in unions (HIP's uint4 has ctors).
struct alignas(16) Q16 { unsigned int w[4]; };

// ---------------------------------------------------------------------------
// Prep 1: enc_proj / dec_proj(+b1). 960 rows total, one block per row.
// ---------------------------------------------------------------------------
__global__ __launch_bounds__(256) void proj_kernel(
    const float* __restrict__ enc, const float* __restrict__ dec,
    const float* __restrict__ W1,  const float* __restrict__ b1,
    float* __restrict__ ep, float* __restrict__ dp)
{
  __shared__ float xs[DE];
  const int r = blockIdx.x;
  const float* x; const float* W; float* out; bool add_b1;
  if (r < BATCH*TT) {                    // encoder rows: W1[0:640]
    x = enc + (size_t)r*DE; W = W1; out = ep + (size_t)r*HID; add_b1 = false;
  } else {                               // decoder rows: W1[640:1280], fold in b1
    const int rd = r - BATCH*TT;
    x = dec + (size_t)rd*DE; W = W1 + (size_t)DE*HID; out = dp + (size_t)rd*HID; add_b1 = true;
  }
  for (int i = threadIdx.x; i < DE; i += 256) xs[i] = x[i];
  __syncthreads();
  for (int col = threadIdx.x; col < HID; col += 256) {
    float acc = add_b1 ? b1[col] : 0.0f;
    #pragma unroll 4
    for (int k = 0; k < DE; ++k) acc = fmaf(xs[k], W[(size_t)k*HID + col], acc);
    out[col] = acc;
  }
}

// ---------------------------------------------------------------------------
// Prep 2: W2 f32 -> bf16, rearranged into WMMA B-fragment order:
//   W2s[kc][ntile][lane][p] , kc=k/32, ntile=n/16,
//   lane = (n%16) + 16*((k%32)>=16), p = k%16  (contiguous k order, same as the A fragment)
// Each lane's 16 bf16 (32B) are contiguous -> two global_load_b128 per frag.
// ---------------------------------------------------------------------------
__global__ __launch_bounds__(256) void shuffle_w2_kernel(
    const float* __restrict__ W2, __bf16* __restrict__ W2s)
{
  const int idx = blockIdx.x*256 + threadIdx.x;   // 0..655359
  const int k = idx >> 10;
  const int n = idx & 1023;
  const int kc = k >> 5, kk = k & 31;
  const int ntile = n >> 4, nl = n & 15;
  const int lane = nl + ((kk & 16) ? 16 : 0);
  const int p = kk & 15;
  const size_t dst = (((size_t)kc*64 + ntile)*32 + lane)*16 + p;
  W2s[dst] = (__bf16)W2[idx];
}

// ---------------------------------------------------------------------------
// Main fused kernel: tanh(ep+dp) staged to LDS as bf16 A tiles, W2s streamed
// from L2 as pre-shuffled B fragments, v_wmma_f32_16x16x32_bf16 accumulation.
// Tile: BM=64 rows ((t,u) pairs) x BN=256 vocab cols; 8 waves as 4x2;
// each wave owns a 16x128 strip = 8 accumulators.
// ---------------------------------------------------------------------------
__global__ __launch_bounds__(256) void joint_kernel(
    const float* __restrict__ ep, const float* __restrict__ dp,
    const __bf16* __restrict__ W2s, const float* __restrict__ b2,
    float* __restrict__ out)
{
  __shared__ alignas(16) __bf16 As[64][40];   // stride 40 elems = 80B (16B-aligned rows)

  const int b   = blockIdx.z;
  const int m0  = blockIdx.x * 64;
  const int n0  = blockIdx.y * 256;
  const int tid  = threadIdx.x;
  const int lane = tid & 31;
  const int wid  = tid >> 5;      // 0..7
  const int wm   = wid >> 1;      // 0..3 : M strip (16 rows)
  const int wn   = wid & 1;       // 0..1 : N strip (128 cols)

  const float* epB = ep + (size_t)b*TT*HID;
  const float* dpB = dp + (size_t)b*UU*HID;

  // A staging: thread -> (row, 8-wide k segment)
  const int srow = tid >> 2;            // 0..63
  const int kseg = (tid & 3) * 8;       // 0,8,16,24
  const int m    = m0 + srow;
  const int t_idx = m / UU;
  const int u_idx = m - t_idx*UU;
  const float* eprow = epB + (size_t)t_idx*HID + kseg;
  const float* dprow = dpB + (size_t)u_idx*HID + kseg;

  const v8f zero = {0.f,0.f,0.f,0.f,0.f,0.f,0.f,0.f};
  v8f acc[8] = {zero, zero, zero, zero, zero, zero, zero, zero};

  // A-fragment LDS address (contiguous k: lanes>=16 start at k=16)
  const int arow = wm*16 + (lane & 15);
  const int akb  = (lane >> 4) * 16;    // element offset: 0 or 16

  const int nt0 = (n0 >> 4) + wn*8;     // first of this wave's 8 n-tiles

  for (int kc = 0; kc < 20; ++kc) {
    const int k0 = kc * 32;
    // ---- stage A tile: hidden = tanh(ep + dp) -> bf16 -> LDS ----
    {
      const float4* e4 = (const float4*)(eprow + k0);
      const float4* d4 = (const float4*)(dprow + k0);
      float4 e0 = e4[0], e1 = e4[1];
      float4 d0 = d4[0], d1 = d4[1];
      union { __bf16 h[8]; Q16 q; } hu;
      hu.h[0] = (__bf16)FAST_TANH(e0.x + d0.x);
      hu.h[1] = (__bf16)FAST_TANH(e0.y + d0.y);
      hu.h[2] = (__bf16)FAST_TANH(e0.z + d0.z);
      hu.h[3] = (__bf16)FAST_TANH(e0.w + d0.w);
      hu.h[4] = (__bf16)FAST_TANH(e1.x + d1.x);
      hu.h[5] = (__bf16)FAST_TANH(e1.y + d1.y);
      hu.h[6] = (__bf16)FAST_TANH(e1.z + d1.z);
      hu.h[7] = (__bf16)FAST_TANH(e1.w + d1.w);
      *(Q16*)&As[srow][kseg] = hu.q;
    }
    __syncthreads();

    // ---- A fragment: two 16B LDS loads (k-runs [akb,akb+8) and [akb+8,akb+16)) ----
    union { Q16 q[2]; v16bf v; } afrag;
    afrag.q[0] = *(const Q16*)&As[arow][akb];
    afrag.q[1] = *(const Q16*)&As[arow][akb + 8];

    // ---- B fragments straight from pre-shuffled global + 8 WMMAs ----
    #pragma unroll
    for (int j = 0; j < 8; ++j) {
      union { Q16 q[2]; v16bf v; } bfrag;
      const Q16* bp = (const Q16*)(W2s + (((size_t)kc*64 + (nt0 + j))*32 + lane)*16);
      bfrag.q[0] = bp[0];
      bfrag.q[1] = bp[1];
      acc[j] = __builtin_amdgcn_wmma_f32_16x16x32_bf16(
          false, afrag.v, false, bfrag.v, (short)0, acc[j], false, false);
    }

    // prefetch next K-chunk's B stream (global_prefetch_b8)
    if (kc < 19) {
      __builtin_prefetch(W2s + (((size_t)(kc+1)*64 + nt0)*32 + lane)*16, 0, 1);
      __builtin_prefetch(W2s + (((size_t)(kc+1)*64 + nt0 + 4)*32 + lane)*16, 0, 1);
    }
    __syncthreads();
  }

  // ---- epilogue: C layout = VGPR i: M=i (lanes 0-15) / M=8+i (lanes 16-31), N=lane&15 ----
  const int mbase = m0 + wm*16 + 8*(lane >> 4);
  const int nbase = n0 + wn*128 + (lane & 15);
  #pragma unroll
  for (int j = 0; j < 8; ++j) {
    const int v = nbase + j*16;
    const float bias = b2[v];
    #pragma unroll
    for (int i = 0; i < 8; ++i) {
      const int mm = mbase + i;
      out[((size_t)b*MBATCH + mm)*VOC + v] = acc[j][i] + bias;
    }
  }
}

// ---------------------------------------------------------------------------
extern "C" void kernel_launch(void* const* d_in, const int* in_sizes, int n_in,
                              void* d_out, int out_size, void* d_ws, size_t ws_size,
                              hipStream_t stream)
{
  const float* enc = (const float*)d_in[0];   // (4,160,640)
  const float* dec = (const float*)d_in[1];   // (4,80,640)
  const float* W1  = (const float*)d_in[2];   // (1280,640)
  const float* b1  = (const float*)d_in[3];   // (640,)
  const float* W2  = (const float*)d_in[4];   // (640,1024)
  const float* b2  = (const float*)d_in[5];   // (1024,)
  float* out = (float*)d_out;                 // (4,160,80,1024)

  char* ws = (char*)d_ws;
  __bf16* W2s = (__bf16*)ws;                          // 1,310,720 B
  float*  ep  = (float*)(ws + 1310720);               // 1,638,400 B
  float*  dp  = (float*)(ws + 1310720 + 1638400);     //   819,200 B
  // total workspace use: ~3.77 MB

  hipLaunchKernelGGL(proj_kernel,       dim3(960),        dim3(256), 0, stream,
                     enc, dec, W1, b1, ep, dp);
  hipLaunchKernelGGL(shuffle_w2_kernel, dim3(2560),       dim3(256), 0, stream,
                     W2, W2s);
  hipLaunchKernelGGL(joint_kernel,      dim3(200, 4, 4),  dim3(256), 0, stream,
                     ep, dp, W2s, b2, out);
}
